// MultiHeadAttention_17617955848580
// MI455X (gfx1250) — matrix operand order repaired and checked by reading, not compiled, not run on hardware
//
#include <hip/hip_runtime.h>
#include <hip/hip_bf16.h>

typedef __attribute__((ext_vector_type(16))) _Float16 v16h;
typedef __attribute__((ext_vector_type(8)))  _Float16 h8;
typedef __attribute__((ext_vector_type(4)))  _Float16 h4;
typedef __attribute__((ext_vector_type(8)))  float    v8f;

#define WMMA_F16(a, b, c) \
  __builtin_amdgcn_wmma_f32_16x16x32_f16(false, (a), false, (b), (short)0, (c), false, false)

static constexpr int BATCH  = 2;
static constexpr int SEQ    = 2048;
static constexpr int DMODEL = 1024;
static constexpr int NHEAD  = 16;
static constexpr int HDIM   = 64;
static constexpr int MROWS  = BATCH * SEQ;   // 4096
static constexpr float SCALE = 0.125f;       // 1/sqrt(64)

// ---------------------------------------------------------------------------
// fp32 -> f16 streaming convert (one pass; removes cvt from all GEMM loops
// and halves L2 read traffic for the matmuls)
// ---------------------------------------------------------------------------
__global__ __launch_bounds__(256) void f32_to_f16_kernel(
    const float* __restrict__ in, _Float16* __restrict__ out, int n)
{
  const int i = (blockIdx.x * 256 + threadIdx.x) * 4;
  if (i < n) {
    const float4 v = *(const float4*)(in + i);
    h4 o;
    o[0] = (_Float16)v.x; o[1] = (_Float16)v.y;
    o[2] = (_Float16)v.z; o[3] = (_Float16)v.w;
    *(h4*)(out + i) = o;
  }
}

// ---------------------------------------------------------------------------
// Fragment loaders (CDNA5 WMMA layouts, ISA 7.12.2):
//   A 16x32 f16 : lane<16 row=lane&15, elems K = {hf*8+i, 16+hf*8+i}
//   B 32x16 f16 : lane<16 col=lane&15, elems K = {hf*8+i, 16+hf*8+i}
//   C 16x16 f32 : VGPR r -> row r + hf*8, col = lane&15
// ---------------------------------------------------------------------------
__device__ __forceinline__ v16h cat16(h8 lo, h8 hi) {
  return __builtin_shufflevector(lo, hi, 0,1,2,3,4,5,6,7,8,9,10,11,12,13,14,15);
}

// 4 A-fragments (rows m0+mt*16, K-slab [k0,k0+32)) from row-major f16 [*, ld]
__device__ __forceinline__ void load_a4(const _Float16* __restrict__ X, int ld,
                                        int m0, int k0, int lane, v16h (&a)[4]) {
  const int hf = lane >> 4, col = lane & 15;
#pragma unroll
  for (int mt = 0; mt < 4; ++mt) {
    const _Float16* p = X + (size_t)(m0 + mt * 16 + col) * ld + k0 + hf * 8;
    a[mt] = cat16(*(const h8*)p, *(const h8*)(p + 16));
  }
}

// 4 B-fragments for B[k][n] = W[n][k]: 8 halves at k0+hf*8 and 8 at +16 per lane
__device__ __forceinline__ void load_b4(const _Float16* __restrict__ W, int ld,
                                        int n0, int k0, int lane, v16h (&b)[4]) {
  const int hf = lane >> 4, col = lane & 15;
#pragma unroll
  for (int nt = 0; nt < 4; ++nt) {
    const _Float16* p = W + (size_t)(n0 + nt * 16 + col) * ld + k0 + hf * 8;
    b[nt] = cat16(*(const h8*)p, *(const h8*)(p + 16));
  }
}

__device__ __forceinline__ void mma16(v8f (&acc)[4][4],
                                      const v16h (&a)[4], const v16h (&b)[4]) {
#pragma unroll
  for (int mt = 0; mt < 4; ++mt)
#pragma unroll
    for (int nt = 0; nt < 4; ++nt)
      acc[mt][nt] = WMMA_F16(a[mt], b[nt], acc[mt][nt]);
}

// ---------------------------------------------------------------------------
// Projection GEMM: Y = X @ W^T + b (all f16 in, f32 accum). One wave computes
// a 64x64 output block (16 WMMAs/k-step, 2-stage pipelined fragment loads).
// Output f16 in head layout [B,H,S,HD], or transposed [B,H,HD,S] for V.
// ---------------------------------------------------------------------------
__global__ __launch_bounds__(128) void proj_qkv_kernel(
    const _Float16* __restrict__ X, const _Float16* __restrict__ W,
    const float* __restrict__ bias, _Float16* __restrict__ Out, int vT)
{
  const int lane = threadIdx.x & 31;
  const int wave = threadIdx.x >> 5;
  const int hf   = lane >> 4;
  const int col  = lane & 15;
  const int m0   = blockIdx.y * 128 + (wave >> 1) * 64;
  const int n0   = blockIdx.x * 128 + (wave & 1) * 64;

  v8f acc[4][4] = {};
  v16h aP[4], bP[4], aQ[4], bQ[4];
  load_a4(X, DMODEL, m0, 0, lane, aP);
  load_b4(W, DMODEL, n0, 0, lane, bP);
  for (int k0 = 0; k0 < DMODEL; k0 += 64) {
    load_a4(X, DMODEL, m0, k0 + 32, lane, aQ);
    load_b4(W, DMODEL, n0, k0 + 32, lane, bQ);
    mma16(acc, aP, bP);
    if (k0 + 64 < DMODEL) {
      load_a4(X, DMODEL, m0, k0 + 64, lane, aP);
      load_b4(W, DMODEL, n0, k0 + 64, lane, bP);
    }
    mma16(acc, aQ, bQ);
  }

#pragma unroll
  for (int nt = 0; nt < 4; ++nt) {
    const int n  = n0 + nt * 16 + col;
    const float bv = bias[n];
    const int hh = n >> 6;
    const int hd = n & (HDIM - 1);
#pragma unroll
    for (int mt = 0; mt < 4; ++mt) {
#pragma unroll
      for (int r = 0; r < 8; ++r) {
        const int m  = m0 + mt * 16 + r + hf * 8;
        const int bb = m >> 11;
        const int s  = m & (SEQ - 1);
        const size_t idx = vT
          ? ((((size_t)(bb * NHEAD + hh)) * HDIM + hd) * SEQ + s)   // V: [B,H,HD,S]
          : ((((size_t)(bb * NHEAD + hh)) * SEQ + s) * HDIM + hd);  // Q,K: [B,H,S,HD]
        Out[idx] = (_Float16)(acc[mt][nt][r] + bv);
      }
    }
  }
}

// ---------------------------------------------------------------------------
// Flash attention, causal. One wave per (b, h, 64-query block): 4 query tiles
// share every loaded K/V fragment (4x less K/V traffic). Scores computed
// TRANSPOSED (S^T = K.Q^T): keys in VGPRs, queries in lanes -> softmax is a
// per-lane reduction + one shfl_xor(16), and exp(S^T) lands directly in the
// A-fragment layout of the P.V WMMA. V is pre-transposed [B,H,HD,S] so its
// B-fragments are two contiguous 8-half runs per lane (keys hf*8.., 16+hf*8..).
// ---------------------------------------------------------------------------
__global__ __launch_bounds__(128) void flash_attn_kernel(
    const _Float16* __restrict__ Qh, const _Float16* __restrict__ Kh,
    const _Float16* __restrict__ Vt, _Float16* __restrict__ AO)
{
  const int lane = threadIdx.x & 31;
  const int wave = threadIdx.x >> 5;
  const int hf   = lane >> 4;
  const int col  = lane & 15;

  const int gid   = blockIdx.x * 4 + wave;
  const int qblk  = gid & 31;                 // S/64 = 32 query blocks
  const int h     = (gid >> 5) & (NHEAD - 1);
  const int b     = gid >> 9;
  const int qbase = qblk * 64;

  const size_t base = (((size_t)b * NHEAD + h) * SEQ) * HDIM;  // same for K and Vt

  // Q as B-fragments for S^T = K.Q^T:  qf[tile][d-chunk]
  v16h qf[4][2];
#pragma unroll
  for (int t = 0; t < 4; ++t) {
    const _Float16* qp = Qh + base + (size_t)(qbase + t * 16 + col) * HDIM;
#pragma unroll
    for (int c = 0; c < 2; ++c) {
      const _Float16* p = qp + c * 32 + hf * 8;
      qf[t][c] = cat16(*(const h8*)p, *(const h8*)(p + 16));
    }
  }

  v8f   acc[4][4] = {};                       // [q-tile][hd-chunk]
  float mrun[4] = {-1e30f, -1e30f, -1e30f, -1e30f};
  float lrun[4] = {0.f, 0.f, 0.f, 0.f};

  for (int kb = 0; kb < qbase + 64; kb += 32) {
    // prefetch next key slab into L2/WGP$ (global_prefetch_b8)
    __builtin_prefetch((const void*)(Kh + base + (size_t)(kb + 32) * HDIM + lane * 8), 0, 3);

    // ---- K A-fragments: ka[key-subtile][d-chunk] ----
    v16h ka[2][2];
#pragma unroll
    for (int st = 0; st < 2; ++st) {
      const _Float16* kp = Kh + base + (size_t)(kb + st * 16 + col) * HDIM;
#pragma unroll
      for (int c = 0; c < 2; ++c) {
        const _Float16* p = kp + c * 32 + hf * 8;
        ka[st][c] = cat16(*(const h8*)p, *(const h8*)(p + 16));
      }
    }
    // ---- V B-fragments (transposed layout): vb[hd-chunk] ----
    v16h vb[4];
#pragma unroll
    for (int cch = 0; cch < 4; ++cch) {
      const _Float16* p = Vt + base + (size_t)(cch * 16 + col) * SEQ + kb + hf * 8;
      vb[cch] = cat16(*(const h8*)p, *(const h8*)(p + 16));
    }

#pragma unroll
    for (int t = 0; t < 4; ++t) {
      const int qb = qbase + t * 16;
      if (kb >= qb + 16) continue;            // wave-uniform causal skip
      const int q = qb + col;                 // this lane's query

      // ---- scores^T: two 16(key)x16(query) tiles, K-dim = HD = 64 ----
      v8f st0 = {}, st1 = {};
      st0 = WMMA_F16(ka[0][0], qf[t][0], st0);
      st0 = WMMA_F16(ka[0][1], qf[t][1], st0);
      st1 = WMMA_F16(ka[1][0], qf[t][0], st1);
      st1 = WMMA_F16(ka[1][1], qf[t][1], st1);

      // ---- scale + causal mask (uniform branch; diagonal step only) ----
      const bool need_mask = (kb + 31) > qb;
#pragma unroll
      for (int r = 0; r < 8; ++r) {
        float s0 = st0[r] * SCALE;
        float s1 = st1[r] * SCALE;
        if (need_mask) {
          const int k0i = kb + r + hf * 8;
          if (k0i > q)      s0 = -1e30f;
          if (k0i + 16 > q) s1 = -1e30f;
        }
        st0[r] = s0;
        st1[r] = s1;
      }

      // ---- online softmax ----
      float mloc = -1e30f;
#pragma unroll
      for (int r = 0; r < 8; ++r) mloc = fmaxf(mloc, fmaxf(st0[r], st1[r]));
      mloc = fmaxf(mloc, __shfl_xor(mloc, 16, 32));
      const float mnew = fmaxf(mrun[t], mloc);
      const float corr = __expf(mrun[t] - mnew);

      v16h pa;                                // P drops straight into A-layout
      float rsum = 0.0f;
#pragma unroll
      for (int r = 0; r < 8; ++r) {
        const float p0 = __expf(st0[r] - mnew);
        const float p1 = __expf(st1[r] - mnew);
        rsum += p0 + p1;
        pa[r]     = (_Float16)p0;             // K = hf*8 + r
        pa[8 + r] = (_Float16)p1;             // K = 16 + hf*8 + r
      }
      rsum += __shfl_xor(rsum, 16, 32);
      lrun[t] = lrun[t] * corr + rsum;
      mrun[t] = mnew;

      // ---- rescale accumulator (row-factor broadcast) ----
#pragma unroll
      for (int r = 0; r < 8; ++r) {
        const float fr = __shfl(corr, r + hf * 8, 32);
        acc[t][0][r] *= fr; acc[t][1][r] *= fr;
        acc[t][2][r] *= fr; acc[t][3][r] *= fr;
      }

      // ---- O += P.V ----
#pragma unroll
      for (int cch = 0; cch < 4; ++cch)
        acc[t][cch] = WMMA_F16(pa, vb[cch], acc[t][cch]);
    }
  }

  // ---- normalize and store f16 merged-head layout [B,S,D] ----
#pragma unroll
  for (int t = 0; t < 4; ++t) {
    const float inv = 1.0f / lrun[t];
#pragma unroll
    for (int r = 0; r < 8; ++r) {
      const float ir = __shfl(inv, r + hf * 8, 32);
      acc[t][0][r] *= ir; acc[t][1][r] *= ir;
      acc[t][2][r] *= ir; acc[t][3][r] *= ir;
    }
#pragma unroll
    for (int cch = 0; cch < 4; ++cch) {
#pragma unroll
      for (int r = 0; r < 8; ++r) {
        const int s = qbase + t * 16 + r + hf * 8;
        const int d = h * HDIM + cch * 16 + col;
        AO[((size_t)(b * SEQ + s)) * DMODEL + d] = (_Float16)acc[t][cch][r];
      }
    }
  }
}

// ---------------------------------------------------------------------------
// Output projection: Y = AO(f16) @ Wo^T + bo -> fp32 d_out, 64x64 per wave
// ---------------------------------------------------------------------------
__global__ __launch_bounds__(128) void out_proj_kernel(
    const _Float16* __restrict__ X, const _Float16* __restrict__ W,
    const float* __restrict__ bias, float* __restrict__ Y)
{
  const int lane = threadIdx.x & 31;
  const int wave = threadIdx.x >> 5;
  const int hf   = lane >> 4;
  const int col  = lane & 15;
  const int m0   = blockIdx.y * 128 + (wave >> 1) * 64;
  const int n0   = blockIdx.x * 128 + (wave & 1) * 64;

  v8f acc[4][4] = {};
  v16h aP[4], bP[4], aQ[4], bQ[4];
  load_a4(X, DMODEL, m0, 0, lane, aP);
  load_b4(W, DMODEL, n0, 0, lane, bP);
  for (int k0 = 0; k0 < DMODEL; k0 += 64) {
    load_a4(X, DMODEL, m0, k0 + 32, lane, aQ);
    load_b4(W, DMODEL, n0, k0 + 32, lane, bQ);
    mma16(acc, aP, bP);
    if (k0 + 64 < DMODEL) {
      load_a4(X, DMODEL, m0, k0 + 64, lane, aP);
      load_b4(W, DMODEL, n0, k0 + 64, lane, bP);
    }
    mma16(acc, aQ, bQ);
  }

#pragma unroll
  for (int nt = 0; nt < 4; ++nt) {
    const int n = n0 + nt * 16 + col;
    const float bv = bias[n];
#pragma unroll
    for (int mt = 0; mt < 4; ++mt) {
#pragma unroll
      for (int r = 0; r < 8; ++r) {
        const int m = m0 + mt * 16 + r + hf * 8;
        Y[(size_t)m * DMODEL + n] = acc[mt][nt][r] + bv;
      }
    }
  }
}

// ---------------------------------------------------------------------------
extern "C" void kernel_launch(void* const* d_in, const int* in_sizes, int n_in,
                              void* d_out, int out_size, void* d_ws, size_t ws_size,
                              hipStream_t stream) {
  (void)in_sizes; (void)n_in; (void)out_size; (void)ws_size;

  const float* q  = (const float*)d_in[0];
  const float* k  = (const float*)d_in[1];
  const float* v  = (const float*)d_in[2];
  // d_in[3] = attn_mask (strictly-upper causal) -- implicit in the kernel
  const float* Wq = (const float*)d_in[4];
  const float* bq = (const float*)d_in[5];
  const float* Wk = (const float*)d_in[6];
  const float* bk = (const float*)d_in[7];
  const float* Wv = (const float*)d_in[8];
  const float* bv = (const float*)d_in[9];
  const float* Wo = (const float*)d_in[10];
  const float* bo = (const float*)d_in[11];
  float* out = (float*)d_out;

  const size_t actElems = (size_t)MROWS * DMODEL;   // 4,194,304
  const size_t wElems   = (size_t)DMODEL * DMODEL;  // 1,048,576

  _Float16* q16  = (_Float16*)d_ws;                 // staged f16 inputs/weights
  _Float16* k16  = q16  + actElems;
  _Float16* v16  = k16  + actElems;
  _Float16* Wq16 = v16  + actElems;
  _Float16* Wk16 = Wq16 + wElems;
  _Float16* Wv16 = Wk16 + wElems;
  _Float16* Wo16 = Wv16 + wElems;
  _Float16* Qh   = Wo16 + wElems;                   // [B,H,S,HD]
  _Float16* Kh   = Qh   + actElems;                 // [B,H,S,HD]
  _Float16* Vt   = Kh   + actElems;                 // [B,H,HD,S] (transposed)
  _Float16* AO   = Vt   + actElems;                 // [B,S,D]

  const dim3 blk(128);

  // stage fp32 -> f16 (single streaming pass, ~96 MiB total)
  const dim3 gCa((actElems / 4 + 255) / 256);
  const dim3 gCw((wElems  / 4 + 255) / 256);
  f32_to_f16_kernel<<<gCa, dim3(256), 0, stream>>>(q,  q16,  (int)actElems);
  f32_to_f16_kernel<<<gCa, dim3(256), 0, stream>>>(k,  k16,  (int)actElems);
  f32_to_f16_kernel<<<gCa, dim3(256), 0, stream>>>(v,  v16,  (int)actElems);
  f32_to_f16_kernel<<<gCw, dim3(256), 0, stream>>>(Wq, Wq16, (int)wElems);
  f32_to_f16_kernel<<<gCw, dim3(256), 0, stream>>>(Wk, Wk16, (int)wElems);
  f32_to_f16_kernel<<<gCw, dim3(256), 0, stream>>>(Wv, Wv16, (int)wElems);
  f32_to_f16_kernel<<<gCw, dim3(256), 0, stream>>>(Wo, Wo16, (int)wElems);

  // Q/K/V projections: 64x64 per wave, 128x128 per block
  const dim3 gProj(DMODEL / 128, MROWS / 128);      // (8, 32)
  proj_qkv_kernel<<<gProj, blk, 0, stream>>>(q16, Wq16, bq, Qh, 0);
  proj_qkv_kernel<<<gProj, blk, 0, stream>>>(k16, Wk16, bk, Kh, 0);
  proj_qkv_kernel<<<gProj, blk, 0, stream>>>(v16, Wv16, bv, Vt, 1);

  // attention: one wave per (b,h,64-query block)
  const dim3 gAttn((BATCH * NHEAD * (SEQ / 64)) / 4);  // 256 blocks
  flash_attn_kernel<<<gAttn, blk, 0, stream>>>(Qh, Kh, Vt, AO);

  out_proj_kernel<<<gProj, blk, 0, stream>>>(AO, Wo16, bo, out);
}
